// Attention_52329881535135
// MI455X (gfx1250) — compile-verified, matrix operand order checked by reading, not run on hardware
//
#include <hip/hip_runtime.h>
#include <hip/hip_bf16.h>

#define RROWS  256
#define SEQ    512
#define DMODEL 256
#define HEADS  8
#define DHEAD  64
#define INNERD 512
#define NTOK   (RROWS * SEQ)   // 131072
#define NEGMAX (-3.402823466e38f)

typedef __attribute__((ext_vector_type(16))) __bf16          v16bf;
typedef __attribute__((ext_vector_type(16))) unsigned short  v16us;
typedef __attribute__((ext_vector_type(8)))  float           v8f;
typedef __attribute__((ext_vector_type(4)))  int             v4i;
typedef __attribute__((ext_vector_type(2)))  __bf16          v2bf;

// Native fptrunc: backend picks the hw f32->bf16 conversion (RNE).
__device__ __forceinline__ unsigned short f2bf(float f) {
  return __builtin_bit_cast(unsigned short, (__bf16)f);
}

// packed f32x2 -> bf16x2 (one dword)
#if defined(__gfx1250__) && __has_builtin(__builtin_amdgcn_cvt_pk_bf16_f32)
__device__ __forceinline__ unsigned pack2bf(float a, float b) {
  return __builtin_bit_cast(unsigned, __builtin_amdgcn_cvt_pk_bf16_f32(a, b));
}
#else
__device__ __forceinline__ unsigned pack2bf(float a, float b) {
  v2bf r = {(__bf16)a, (__bf16)b};
  return __builtin_bit_cast(unsigned, r);
}
#endif

// ---- async global->LDS (CDNA5 GLOBAL_LOAD_ASYNC_TO_LDS_B128, ASYNCcnt) ----
#if defined(__gfx1250__) && __has_builtin(__builtin_amdgcn_global_load_async_to_lds_b128)
#define HAVE_ASYNC 1
typedef __attribute__((address_space(1))) v4i* async_gptr;
typedef __attribute__((address_space(3))) v4i* async_lptr;
__device__ __forceinline__ void async_cp16(unsigned short* l, const unsigned short* g) {
  __builtin_amdgcn_global_load_async_to_lds_b128(
      (async_gptr)(unsigned long long)(uintptr_t)g,
      (async_lptr)(unsigned)(uintptr_t)l, 0, 0);
}
__device__ __forceinline__ void async_wait0() { asm volatile("s_wait_asynccnt 0" ::: "memory"); }
__device__ __forceinline__ void async_wait4() { asm volatile("s_wait_asynccnt 4" ::: "memory"); }
#else
#define HAVE_ASYNC 0
__device__ __forceinline__ void async_cp16(unsigned short* l, const unsigned short* g) {
  *(uint4*)l = *(const uint4*)g;
}
__device__ __forceinline__ void async_wait0() {}
__device__ __forceinline__ void async_wait4() {}
#endif

// Fragment-major LDS layout for a [rows x 32] bf16 tile (A-role; B-role uses
// n in place of m). Element (m,k) lives at lane=(m&15)+16*((k>>3)&1),
// elem=(k&7)+8*(k>>4) of strip m>>4 — 16 contiguous bf16 per lane.
__device__ __forceinline__ int fragidx(int mn, int k) {
  return (((mn >> 4) * 32 + (mn & 15) + (((k >> 3) & 1) << 4)) << 4) +
         (k & 7) + ((k >> 4) << 3);
}

// One aligned 32B vector load -> full 16-element fragment, no shuffles.
__device__ __forceinline__ v16bf frag_ld(const unsigned short* p, int strip, int lane) {
  return __builtin_bit_cast(v16bf, *(const v16us*)(p + ((strip * 32 + lane) << 4)));
}

__device__ __forceinline__ v8f wmma_bf16(v16bf a, v16bf b, v8f c) {
  return __builtin_amdgcn_wmma_f32_16x16x32_bf16(false, a, false, b, (short)0, c,
                                                 false, false);
}

// ---------------------------------------------------------------- mask stats
__global__ __launch_bounds__(512) void mask_stats_kernel(
    const unsigned char* __restrict__ mask, float* __restrict__ rowflag,
    float* __restrict__ numrows, float* __restrict__ maskany) {
  __shared__ float sflag[RROWS];
  const int t = threadIdx.x;
  if (t < RROWS) {
    const unsigned char* mrow = mask + (size_t)t * SEQ;
    int any = 0;
    for (int j = 0; j < SEQ; ++j) any |= mrow[j];
    float f = any ? 1.0f : 0.0f;
    rowflag[t] = f;
    sflag[t] = f;
  }
  __syncthreads();
  if (t == 0) {
    float s = 0.0f;
    for (int i = 0; i < RROWS; ++i) s += sflag[i];
    numrows[0] = s;
  }
  int anyc = 0;
  for (int r = 0; r < RROWS; ++r) anyc |= mask[(size_t)r * SEQ + t];
  maskany[t] = anyc ? 1.0f : 0.0f;
}

// ------------------------------------------------- q/k/v projection (z=0,1,2)
__global__ __launch_bounds__(256) void proj_kernel(
    const float* __restrict__ x, const float* __restrict__ Wq,
    const float* __restrict__ Wk, const float* __restrict__ Wv,
    const float* __restrict__ rowflag, unsigned short* __restrict__ qb,
    unsigned short* __restrict__ kb, unsigned short* __restrict__ vb) {
  __shared__ __align__(32) unsigned short As[128 * 32];
  __shared__ __align__(32) unsigned short Bs[32 * 64];
  const int tid = threadIdx.x, lane = tid & 31, w = tid >> 5;
  const int M0 = blockIdx.x * 128, N0 = blockIdx.y * 64, z = blockIdx.z;
  const float* W = (z == 0) ? Wq : (z == 1) ? Wk : Wv;
  unsigned short* dst = (z == 0) ? qb : (z == 1) ? kb : vb;
  const float flag = (z == 0) ? rowflag[M0 >> 9] : 1.0f;  // 128 | 512

  v8f acc[4];
#pragma unroll
  for (int nt = 0; nt < 4; ++nt) acc[nt] = (v8f){0, 0, 0, 0, 0, 0, 0, 0};

  const int arow = tid >> 1, acb = (tid & 1) * 16;  // A: 128 x 32 f32
  const int bn = tid & 63, bkg = (tid >> 6) * 8;    // B: 32 x 64 f32 (k-strided)

  for (int k0 = 0; k0 < DMODEL; k0 += 32) {
    const float* ap = x + (size_t)(M0 + arow) * DMODEL + k0 + acb;
    unsigned* d0 = (unsigned*)&As[fragidx(arow, acb)];
    unsigned* d1 = (unsigned*)&As[fragidx(arow, acb + 8)];
#pragma unroll
    for (int j = 0; j < 4; ++j) d0[j] = pack2bf(ap[2 * j], ap[2 * j + 1]);
#pragma unroll
    for (int j = 0; j < 4; ++j) d1[j] = pack2bf(ap[8 + 2 * j], ap[9 + 2 * j]);
    const float* bp = W + (size_t)(k0 + bkg) * INNERD + N0 + bn;
    unsigned* db = (unsigned*)&Bs[fragidx(bn, bkg)];
#pragma unroll
    for (int j = 0; j < 4; ++j)
      db[j] = pack2bf(bp[(2 * j) * INNERD], bp[(2 * j + 1) * INNERD]);
    __syncthreads();
    v16bf a = frag_ld(As, w, lane);
#pragma unroll
    for (int nt = 0; nt < 4; ++nt)
      acc[nt] = wmma_bf16(a, frag_ld(Bs, nt, lane), acc[nt]);
    __syncthreads();
  }
  const int lm = (lane >> 4) << 3, ln = lane & 15;
#pragma unroll
  for (int nt = 0; nt < 4; ++nt)
#pragma unroll
    for (int e = 0; e < 8; ++e) {
      int t = M0 + w * 16 + lm + e;
      dst[(size_t)t * INNERD + N0 + nt * 16 + ln] = f2bf(acc[nt][e] * flag);
    }
}

// ------------------------------------- row-tied logits: sum_r Q_r K_r^T / s
__device__ __forceinline__ void logits_issue(
    const unsigned short* qb, const unsigned short* kbuf, unsigned short* Qs,
    unsigned short* Ks, int it, int I0, int J0, size_t hoff, int qrow, int qcb,
    int kj, int kkg) {
  const int r = it >> 1, dc = it & 1;
  const size_t koff = hoff + dc * 32;
  const unsigned short* qp =
      qb + (size_t)(r * SEQ + I0 + qrow) * INNERD + koff + qcb;
  async_cp16(&Qs[fragidx(qrow, qcb)], qp);
  async_cp16(&Qs[fragidx(qrow, qcb + 8)], qp + 8);
  // K^T tile: source (row j, 8 contiguous d) <-> dest contiguous fragment run
  const unsigned short* kp =
      kbuf + (size_t)(r * SEQ + J0 + kj) * INNERD + koff + kkg;
  async_cp16(&Ks[fragidx(kj, kkg)], kp);
  async_cp16(&Ks[fragidx(kj, kkg + 16)], kp + 16);
}

__global__ __launch_bounds__(256) void logits_kernel(
    const unsigned short* __restrict__ qb, const unsigned short* __restrict__ kbuf,
    const float* __restrict__ numrows, float* __restrict__ logits) {
  __shared__ __align__(32) unsigned short Qs[2][128 * 32];
  __shared__ __align__(32) unsigned short Ks[2][128 * 32];
  const int tid = threadIdx.x, lane = tid & 31, w = tid >> 5;
  const int I0 = blockIdx.x * 128, J0 = blockIdx.y * 128, h = blockIdx.z;
  const int wm = w & 3, wn = w >> 2;
  const size_t hoff = (size_t)h * DHEAD;
  const float scale = 0.125f * rsqrtf(numrows[0]);  // DH^-0.5 * num_rows^-0.5

  v8f acc[2][4];
#pragma unroll
  for (int s = 0; s < 2; ++s)
#pragma unroll
    for (int nt = 0; nt < 4; ++nt) acc[s][nt] = (v8f){0, 0, 0, 0, 0, 0, 0, 0};

  const int qrow = tid >> 1, qcb = (tid & 1) * 16;
  const int kj = tid & 127, kkg = (tid >> 7) * 8;

  logits_issue(qb, kbuf, Qs[0], Ks[0], 0, I0, J0, hoff, qrow, qcb, kj, kkg);
  for (int it = 0; it < 2 * RROWS; ++it) {
    const int cur = it & 1;
    if (it + 1 < 2 * RROWS) {
      logits_issue(qb, kbuf, Qs[cur ^ 1], Ks[cur ^ 1], it + 1, I0, J0, hoff,
                   qrow, qcb, kj, kkg);
      async_wait4();  // retire current buffer's 4 transfers only
    } else {
      async_wait0();
    }
    __syncthreads();
    v16bf a0 = frag_ld(Qs[cur], wm * 2, lane);
    v16bf a1 = frag_ld(Qs[cur], wm * 2 + 1, lane);
#pragma unroll
    for (int nt = 0; nt < 4; ++nt) {
      v16bf b = frag_ld(Ks[cur], wn * 4 + nt, lane);
      acc[0][nt] = wmma_bf16(a0, b, acc[0][nt]);
      acc[1][nt] = wmma_bf16(a1, b, acc[1][nt]);
    }
    __syncthreads();
  }
  const int lm = (lane >> 4) << 3, ln = lane & 15;
#pragma unroll
  for (int s = 0; s < 2; ++s)
#pragma unroll
    for (int nt = 0; nt < 4; ++nt)
#pragma unroll
      for (int e = 0; e < 8; ++e) {
        int i = I0 + wm * 32 + s * 16 + lm + e;
        int j = J0 + wn * 64 + nt * 16 + ln;
        logits[((size_t)h * SEQ + i) * SEQ + j] = acc[s][nt][e] * scale;
      }
}

// --------------------------------------------------- masked softmax -> bf16
__global__ __launch_bounds__(256) void softmax_kernel(
    const float* __restrict__ logits, const float* __restrict__ maskany,
    unsigned short* __restrict__ attn) {
  __shared__ float red[256];
  const int i = blockIdx.x, h = blockIdx.y, t = threadIdx.x;
  const float* row = logits + ((size_t)h * SEQ + i) * SEQ;
  const float mi = maskany[i];
  float v0 = (mi > 0.0f && maskany[t] > 0.0f) ? row[t] : NEGMAX;
  float v1 = (mi > 0.0f && maskany[t + 256] > 0.0f) ? row[t + 256] : NEGMAX;
  red[t] = fmaxf(v0, v1);
  __syncthreads();
  for (int s = 128; s > 0; s >>= 1) {
    if (t < s) red[t] = fmaxf(red[t], red[t + s]);
    __syncthreads();
  }
  const float mx = red[0];
  __syncthreads();
  const float e0 = __expf(v0 - mx), e1 = __expf(v1 - mx);
  red[t] = e0 + e1;
  __syncthreads();
  for (int s = 128; s > 0; s >>= 1) {
    if (t < s) red[t] += red[t + s];
    __syncthreads();
  }
  const float inv = 1.0f / red[0];
  unsigned short* arow = attn + ((size_t)h * SEQ + i) * SEQ;
  arow[t] = f2bf(e0 * inv);
  arow[t + 256] = f2bf(e1 * inv);
}

// ------------------------------------ out_pre[r,i,h,:] = attn[h] @ V[r,:,h,:]
__global__ __launch_bounds__(256) void av_kernel(
    const unsigned short* __restrict__ attn, const unsigned short* __restrict__ vb,
    unsigned short* __restrict__ outp) {
  __shared__ __align__(32) unsigned short As[128 * 32];
  __shared__ __align__(32) unsigned short Bs[32 * 64];
  const int tid = threadIdx.x, lane = tid & 31, w = tid >> 5;
  const int I0 = blockIdx.x * 128, r = blockIdx.y, h = blockIdx.z;

  v8f acc[4];
#pragma unroll
  for (int nt = 0; nt < 4; ++nt) acc[nt] = (v8f){0, 0, 0, 0, 0, 0, 0, 0};

  const int arow = tid >> 1, acb = (tid & 1) * 16;
  const int bn = tid & 63, bkg = (tid >> 6) * 8;

  for (int kt = 0; kt < 16; ++kt) {
    const int j0 = kt * 32;
    const unsigned short* ap =
        attn + ((size_t)h * SEQ + I0 + arow) * SEQ + j0 + acb;
    async_cp16(&As[fragidx(arow, acb)], ap);
    async_cp16(&As[fragidx(arow, acb + 8)], ap + 8);
    const unsigned short* bp =
        vb + (size_t)(r * SEQ + j0 + bkg) * INNERD + h * DHEAD + bn;
    unsigned* db = (unsigned*)&Bs[fragidx(bn, bkg)];
#pragma unroll
    for (int j = 0; j < 4; ++j)
      db[j] = (unsigned)bp[(2 * j) * INNERD] |
              ((unsigned)bp[(2 * j + 1) * INNERD] << 16);
    async_wait0();
    __syncthreads();
    v16bf a = frag_ld(As, w, lane);
#pragma unroll
    for (int nt = 0; nt < 4; ++nt)
      acc[nt] = wmma_bf16(a, frag_ld(Bs, nt, lane), acc[nt]);
    __syncthreads();
  }
  const int lm = (lane >> 4) << 3, ln = lane & 15;
#pragma unroll
  for (int nt = 0; nt < 4; ++nt)
#pragma unroll
    for (int e = 0; e < 8; ++e) {
      int t = r * SEQ + I0 + w * 16 + lm + e;
      outp[(size_t)t * INNERD + h * DHEAD + nt * 16 + ln] = f2bf(acc[nt][e]);
    }
}

// ------------------------------------------------ out = out_pre @ Wo + bo
__global__ __launch_bounds__(256) void oproj_kernel(
    const unsigned short* __restrict__ outp, const float* __restrict__ Wo,
    const float* __restrict__ bo, float* __restrict__ out) {
  __shared__ __align__(32) unsigned short As[128 * 32];
  __shared__ __align__(32) unsigned short Bs[32 * 64];
  const int tid = threadIdx.x, lane = tid & 31, w = tid >> 5;
  const int M0 = blockIdx.x * 128, N0 = blockIdx.y * 64;

  v8f acc[4];
#pragma unroll
  for (int nt = 0; nt < 4; ++nt) acc[nt] = (v8f){0, 0, 0, 0, 0, 0, 0, 0};

  const int arow = tid >> 1, acb = (tid & 1) * 16;
  const int bn = tid & 63, bkg = (tid >> 6) * 8;

  for (int k0 = 0; k0 < INNERD; k0 += 32) {
    const unsigned short* ap = outp + (size_t)(M0 + arow) * INNERD + k0 + acb;
    async_cp16(&As[fragidx(arow, acb)], ap);
    async_cp16(&As[fragidx(arow, acb + 8)], ap + 8);
    const float* bp = Wo + (size_t)(k0 + bkg) * DMODEL + N0 + bn;
    unsigned* db = (unsigned*)&Bs[fragidx(bn, bkg)];
#pragma unroll
    for (int j = 0; j < 4; ++j)
      db[j] = pack2bf(bp[(2 * j) * DMODEL], bp[(2 * j + 1) * DMODEL]);
    async_wait0();
    __syncthreads();
    v16bf a = frag_ld(As, w, lane);
#pragma unroll
    for (int nt = 0; nt < 4; ++nt)
      acc[nt] = wmma_bf16(a, frag_ld(Bs, nt, lane), acc[nt]);
    __syncthreads();
  }
  const int lm = (lane >> 4) << 3, ln = lane & 15;
#pragma unroll
  for (int nt = 0; nt < 4; ++nt)
#pragma unroll
    for (int e = 0; e < 8; ++e) {
      int t = M0 + w * 16 + lm + e;
      int n = N0 + nt * 16 + ln;
      out[(size_t)t * DMODEL + n] = acc[nt][e] + bo[n];
    }
}

extern "C" void kernel_launch(void* const* d_in, const int* in_sizes, int n_in,
                              void* d_out, int out_size, void* d_ws, size_t ws_size,
                              hipStream_t stream) {
  (void)in_sizes; (void)n_in; (void)out_size; (void)ws_size;
  const float* x            = (const float*)d_in[0];
  const unsigned char* mask = (const unsigned char*)d_in[1];
  const float* Wq           = (const float*)d_in[2];
  const float* Wk           = (const float*)d_in[3];
  const float* Wv           = (const float*)d_in[4];
  const float* Wo           = (const float*)d_in[5];
  const float* bo           = (const float*)d_in[6];
  float* out                = (float*)d_out;

  char* ws = (char*)d_ws;
  const size_t SZ_QKV = (size_t)NTOK * INNERD * sizeof(unsigned short);  // 128 MiB
  unsigned short* qb = (unsigned short*)ws;
  unsigned short* kb = (unsigned short*)(ws + SZ_QKV);
  unsigned short* vb = (unsigned short*)(ws + 2 * SZ_QKV);
  float* logits      = (float*)(ws + 3 * SZ_QKV);
  float* stats       = (float*)(ws + 3 * SZ_QKV +
                                (size_t)HEADS * SEQ * SEQ * sizeof(float));
  float* rowflag = stats;
  float* numrows = stats + 256;
  float* maskany = stats + 272;
  unsigned short* attn = qb;  // q dead after logits_kernel
  unsigned short* outp = kb;  // k dead after logits_kernel

  mask_stats_kernel<<<1, 512, 0, stream>>>(mask, rowflag, numrows, maskany);
  proj_kernel<<<dim3(NTOK / 128, INNERD / 64, 3), 256, 0, stream>>>(
      x, Wq, Wk, Wv, rowflag, qb, kb, vb);
  logits_kernel<<<dim3(SEQ / 128, SEQ / 128, HEADS), 256, 0, stream>>>(
      qb, kb, numrows, logits);
  softmax_kernel<<<dim3(SEQ, HEADS), 256, 0, stream>>>(logits, maskany, attn);
  av_kernel<<<dim3(SEQ / 128, RROWS, HEADS), 256, 0, stream>>>(attn, vb, outp);
  oproj_kernel<<<dim3(NTOK / 128, DMODEL / 64), 256, 0, stream>>>(outp, Wo, bo, out);
}
